// JumpReactionDiffusionODE_90872918049134
// MI455X (gfx1250) — matrix operand order repaired and checked by reading, not compiled, not run on hardware
//
#include <hip/hip_runtime.h>
#include <hip/hip_bf16.h>
#include <math.h>

typedef __attribute__((ext_vector_type(16))) __bf16 v16bf;
typedef __attribute__((ext_vector_type(8)))  float  v8f;

#define D_MODEL    64
#define K_HEADS    8
#define D_HEAD     8
#define T_LEN      12
#define C_IN       3
#define TILE_N     16
#define ROWS       (T_LEN * TILE_N)   // 192 rows per tile
#define THREADS    128                // 4 waves (wave32)
#define NRD        2
#define RESOLUTION 10

// ---------------------------------------------------------------------------
// Dense phase: fully fused branch (fc2 -> temporal attention -> fc2 -> head)
// ---------------------------------------------------------------------------

struct BranchParams {
  const float* inputs;                      // (B, C, N, T)
  const float *w_in1, *b_in1, *w_in2, *b_in2;
  const float *w_qkv, *b_qkv;               // (3,64,64), (3,64)
  const float *w_a1, *b_a1, *w_a2, *b_a2;
  const float *w_out1, *b_out1, *w_out2, *b_out2;
  float* out;                               // (B, T, N, cout)
  int N, B, cout;
};

__device__ inline void load_weight_lds(const float* wg, const float* bg,
                                       __bf16 (*sw)[D_MODEL], float* sb, int tid) {
  for (int i = tid; i < D_MODEL * D_MODEL; i += THREADS)
    sw[i >> 6][i & 63] = (__bf16)wg[i];
  if (tid < D_MODEL) sb[tid] = bg[tid];
  __syncthreads();
}

// out = act(in @ sw^T + sb), in/out: [ROWS][64] bf16 in LDS, sw: [d_out][c_in]
__device__ inline void gemm_wmma(const __bf16 (*in)[D_MODEL], __bf16 (*out)[D_MODEL],
                                 const __bf16 (*sw)[D_MODEL], const float* sb,
                                 int tid, bool relu) {
  const int lane = tid & 31;
  const int wave = tid >> 5;
  const int ml   = lane & 15;   // row / column within tile
  const int kh   = lane >> 4;   // which K-half this lane holds
  for (int m0 = wave; m0 < ROWS / 16; m0 += THREADS / 32) {
    // A fragments (16x32 bf16): lane<16 holds K 0..7,16..23; lane>=16 holds 8..15,24..31
    v16bf a0, a1;
    const int row = m0 * 16 + ml;
    #pragma unroll
    for (int r = 0; r < 8; ++r) {
      const int kk = (r < 4) ? (kh * 8 + 2 * r) : (16 + kh * 8 + 2 * (r - 4));
      a0[2 * r]     = in[row][kk];
      a0[2 * r + 1] = in[row][kk + 1];
      a1[2 * r]     = in[row][32 + kk];
      a1[2 * r + 1] = in[row][32 + kk + 1];
    }
    #pragma unroll
    for (int n0 = 0; n0 < 4; ++n0) {
      const int nc = n0 * 16 + ml;  // output column (lane holds N = lane%16)
      // B fragments (32x16 bf16): B[k][n] = w[n][k] -> contiguous K pairs in sw[nc]
      v16bf b0, b1;
      #pragma unroll
      for (int r = 0; r < 8; ++r) {
        const int kk = (r < 4) ? (kh * 8 + 2 * r) : (16 + kh * 8 + 2 * (r - 4));
        b0[2 * r]     = sw[nc][kk];
        b0[2 * r + 1] = sw[nc][kk + 1];
        b1[2 * r]     = sw[nc][32 + kk];
        b1[2 * r + 1] = sw[nc][32 + kk + 1];
      }
      v8f acc = {};
      acc = __builtin_amdgcn_wmma_f32_16x16x32_bf16(false, a0, false, b0,
                                                    (short)0, acc, false, false);
      acc = __builtin_amdgcn_wmma_f32_16x16x32_bf16(false, a1, false, b1,
                                                    (short)0, acc, false, false);
      const float bs = sb[nc];
      #pragma unroll
      for (int r = 0; r < 8; ++r) {   // C/D layout: M = r + 8*(lane/16)
        float v = acc[r] + bs;
        if (relu) v = fmaxf(v, 0.0f);
        out[m0 * 16 + 8 * kh + r][nc] = (__bf16)v;
      }
    }
  }
  __syncthreads();
}

__global__ __launch_bounds__(THREADS)
void branch_kernel(BranchParams p) {
  __shared__ __bf16 s_cur[ROWS][D_MODEL];
  __shared__ __bf16 s_nxt[ROWS][D_MODEL];
  __shared__ __bf16 s_q[ROWS][D_MODEL];
  __shared__ __bf16 s_k[ROWS][D_MODEL];
  __shared__ __bf16 s_v[ROWS][D_MODEL];
  __shared__ __bf16 s_w[D_MODEL][D_MODEL];
  __shared__ float  s_bias[D_MODEL];
  __shared__ float  s_x[ROWS][4];

  const int tid   = threadIdx.x;
  const int N     = p.N;
  const int tiles = N / TILE_N;
  const int b     = blockIdx.x / tiles;
  const int node0 = (blockIdx.x % tiles) * TILE_N;

  // Load input tile: x[row=t*16+nl][c] = inputs[b][c][node0+nl][t]
  for (int i = tid; i < ROWS * C_IN; i += THREADS) {
    const int row = i / C_IN, c = i % C_IN;
    const int t = row >> 4, nl = row & 15;
    s_x[row][c] = p.inputs[(((size_t)b * C_IN + c) * N + node0 + nl) * T_LEN + t];
  }
  __syncthreads();

  // Layer in1 (K=3) + relu: too skinny for WMMA, do on VALU
  for (int i = tid; i < ROWS * D_MODEL; i += THREADS) {
    const int row = i >> 6, d = i & 63;
    float s = p.b_in1[d];
    #pragma unroll
    for (int c = 0; c < C_IN; ++c) s += s_x[row][c] * p.w_in1[d * C_IN + c];
    s_cur[row][d] = (__bf16)fmaxf(s, 0.0f);
  }
  __syncthreads();

  // in2 (linear)
  load_weight_lds(p.w_in2, p.b_in2, s_w, s_bias, tid);
  gemm_wmma(s_cur, s_nxt, s_w, s_bias, tid, false);

  // qkv (relu on each)
  load_weight_lds(p.w_qkv + 0 * D_MODEL * D_MODEL, p.b_qkv + 0 * D_MODEL, s_w, s_bias, tid);
  gemm_wmma(s_nxt, s_q, s_w, s_bias, tid, true);
  load_weight_lds(p.w_qkv + 1 * D_MODEL * D_MODEL, p.b_qkv + 1 * D_MODEL, s_w, s_bias, tid);
  gemm_wmma(s_nxt, s_k, s_w, s_bias, tid, true);
  load_weight_lds(p.w_qkv + 2 * D_MODEL * D_MODEL, p.b_qkv + 2 * D_MODEL, s_w, s_bias, tid);
  gemm_wmma(s_nxt, s_v, s_w, s_bias, tid, true);

  // Temporal attention: one (node, head) per thread, T=12, D_HEAD=8
  {
    const int nl = tid & 15;
    const int h  = tid >> 4;                    // 0..7
    const float scale = 0.35355339059327373f;   // 1/sqrt(D_HEAD)
    for (int t = 0; t < T_LEN; ++t) {
      float qv[D_HEAD];
      #pragma unroll
      for (int d = 0; d < D_HEAD; ++d)
        qv[d] = (float)s_q[t * 16 + nl][h * D_HEAD + d];
      float sc[T_LEN];
      float mx = -3.4e38f;
      for (int s2 = 0; s2 < T_LEN; ++s2) {
        float dot = 0.f;
        #pragma unroll
        for (int d = 0; d < D_HEAD; ++d)
          dot += qv[d] * (float)s_k[s2 * 16 + nl][h * D_HEAD + d];
        dot *= scale;
        sc[s2] = dot;
        mx = fmaxf(mx, dot);
      }
      float sum = 0.f;
      for (int s2 = 0; s2 < T_LEN; ++s2) { float e = __expf(sc[s2] - mx); sc[s2] = e; sum += e; }
      const float inv = 1.0f / sum;
      float o[D_HEAD] = {};
      for (int s2 = 0; s2 < T_LEN; ++s2) {
        const float pr = sc[s2] * inv;
        #pragma unroll
        for (int d = 0; d < D_HEAD; ++d)
          o[d] += pr * (float)s_v[s2 * 16 + nl][h * D_HEAD + d];
      }
      #pragma unroll
      for (int d = 0; d < D_HEAD; ++d)
        s_cur[t * 16 + nl][h * D_HEAD + d] = (__bf16)o[d];
    }
  }
  __syncthreads();

  // a1 (relu), a2 (linear)
  load_weight_lds(p.w_a1, p.b_a1, s_w, s_bias, tid);
  gemm_wmma(s_cur, s_nxt, s_w, s_bias, tid, true);
  load_weight_lds(p.w_a2, p.b_a2, s_w, s_bias, tid);
  gemm_wmma(s_nxt, s_cur, s_w, s_bias, tid, false);

  // out1 (relu)
  load_weight_lds(p.w_out1, p.b_out1, s_w, s_bias, tid);
  gemm_wmma(s_cur, s_nxt, s_w, s_bias, tid, true);

  // out2: (cout x 64), cout<=2 -> scalar dot products, write to global
  const int cout = p.cout;
  for (int i = tid; i < ROWS * cout; i += THREADS) {
    const int row = i / cout, co = i % cout;
    float s = p.b_out2[co];
    for (int c = 0; c < D_MODEL; ++c)
      s += (float)s_nxt[row][c] * p.w_out2[co * D_MODEL + c];
    const int t = row >> 4, nl = row & 15;
    p.out[(((size_t)b * T_LEN + t) * N + node0 + nl) * cout + co] = s;
  }
}

// ---------------------------------------------------------------------------
// ODE phase kernels (L2-resident sparse diffusion)
// ---------------------------------------------------------------------------

__global__ void softmax2_kernel(float* g, int n) {
  const int i = blockIdx.x * blockDim.x + threadIdx.x;
  if (i >= n) return;
  const float a = g[2 * i], b = g[2 * i + 1];
  const float m = fmaxf(a, b);
  const float ea = __expf(a - m), eb = __expf(b - m);
  const float inv = 1.0f / (ea + eb);
  g[2 * i] = ea * inv;
  g[2 * i + 1] = eb * inv;
}

__global__ void ode_init_kernel(const float* __restrict__ inputs, float* __restrict__ xs,
                                float* __restrict__ msg, float* __restrict__ wdeg,
                                int B, int N, int T) {
  const int i = blockIdx.x * blockDim.x + threadIdx.x;
  if (i < NRD * B * N) msg[i] = 0.f;
  if (i < NRD * N) wdeg[i] = 0.f;
  if (i < B * N) {
    const int b = i / N, n = i % N;
    xs[i] = inputs[(((size_t)b * C_IN + 0) * N + n) * T + (T - 1)];
  }
}

__global__ void wdeg_kernel(const float* __restrict__ edge_w, const int* __restrict__ dst,
                            float* __restrict__ wdeg, int E, int N) {
  const int e = blockIdx.x * blockDim.x + threadIdx.x;
  if (e >= E) return;
  const int d = dst[e];
  #pragma unroll
  for (int k = 0; k < NRD; ++k)
    atomicAdd(&wdeg[k * N + d], edge_w[(size_t)k * E + e]);
}

__global__ void edge_kernel(const float* __restrict__ xs, const float* __restrict__ edge_w,
                            const int* __restrict__ src, const int* __restrict__ dst,
                            float* __restrict__ msg, int E, int N, int B) {
  const int e = blockIdx.x * blockDim.x + threadIdx.x;
  if (e >= E) return;
  const int s = src[e], d = dst[e];
  #pragma unroll
  for (int k = 0; k < NRD; ++k) {
    const float w = edge_w[(size_t)k * E + e];
    for (int b = 0; b < B; ++b)
      atomicAdd(&msg[((size_t)k * B + b) * N + d], xs[b * N + s] * w);
  }
}

__global__ void node_kernel(float* __restrict__ xs, float* __restrict__ msg,
                            const float* __restrict__ wdeg,
                            const float* __restrict__ l, const float* __restrict__ jt,
                            const float* __restrict__ diff_a, const float* __restrict__ react_r,
                            int B, int N, int T, int step, float dt) {
  const int i = blockIdx.x * blockDim.x + threadIdx.x;
  if (i >= B * N) return;
  const int b = i / N, n = i % N;
  const int Tg = T - 1;
  int idx = step / RESOLUTION;
  if (idx > Tg - 1) idx = Tg - 1;
  float x = xs[i];
  const size_t tb = ((size_t)b * T + idx) * N + n;
  const float lk0 = l[2 * tb], lk1 = l[2 * tb + 1];
  float f = jt[tb];
  const float lk[NRD] = {lk0, lk1};
  #pragma unroll
  for (int k = 0; k < NRD; ++k) {
    const size_t mi = ((size_t)k * B + b) * N + n;
    const float mg = msg[mi];
    msg[mi] = 0.f;                                  // reset for next step
    const float diff = mg - x * wdeg[k * N + n];
    const float rd = diff_a[k] * diff + react_r[k] * x * (1.f - x);
    f += lk[k] * rd;
  }
  xs[i] = x + dt * f;
}

// ---------------------------------------------------------------------------
// Launch
// ---------------------------------------------------------------------------

extern "C" void kernel_launch(void* const* d_in, const int* in_sizes, int n_in,
                              void* d_out, int out_size, void* d_ws, size_t ws_size,
                              hipStream_t stream) {
  const float* inputs  = (const float*)d_in[0];
  const float* edge_w  = (const float*)d_in[1];
  const float* diff_a  = (const float*)d_in[2];
  const float* react_r = (const float*)d_in[3];
  const float* gw[14];
  const float* jw[14];
  for (int i = 0; i < 14; ++i) gw[i] = (const float*)d_in[4 + i];
  for (int i = 0; i < 14; ++i) jw[i] = (const float*)d_in[18 + i];
  const int* src = (const int*)d_in[32];
  const int* dst = (const int*)d_in[33];

  const int B = 4;
  const int T = T_LEN;
  const int N = in_sizes[0] / (B * C_IN * T);   // 10000
  const int E = in_sizes[32];                   // 160000

  // Workspace layout (floats)
  float* ws   = (float*)d_ws;
  float* gate = ws;                                 // B*T*N*2 (raw -> softmax l)
  float* jt   = gate + (size_t)B * T * N * 2;       // B*T*N
  float* msg  = jt + (size_t)B * T * N;             // NRD*B*N
  float* wdeg = msg + (size_t)NRD * B * N;          // NRD*N
  float* xs   = (float*)d_out;                      // B*N, integrated in place

  BranchParams pg = {inputs, gw[0], gw[1], gw[2], gw[3], gw[4], gw[5], gw[6],
                     gw[7], gw[8], gw[9], gw[10], gw[11], gw[12], gw[13],
                     gate, N, B, 2};
  BranchParams pj = {inputs, jw[0], jw[1], jw[2], jw[3], jw[4], jw[5], jw[6],
                     jw[7], jw[8], jw[9], jw[10], jw[11], jw[12], jw[13],
                     jt, N, B, 1};

  const dim3 bgrid(B * (N / TILE_N));
  branch_kernel<<<bgrid, THREADS, 0, stream>>>(pg);
  branch_kernel<<<bgrid, THREADS, 0, stream>>>(pj);

  const int nSoft = B * T * N;
  softmax2_kernel<<<(nSoft + 255) / 256, 256, 0, stream>>>(gate, nSoft);

  const int nInit = NRD * B * N;  // covers msg, wdeg, xs ranges
  ode_init_kernel<<<(nInit + 255) / 256, 256, 0, stream>>>(inputs, xs, msg, wdeg, B, N, T);
  wdeg_kernel<<<(E + 255) / 256, 256, 0, stream>>>(edge_w, dst, wdeg, E, N);

  const int Tg = T - 1;
  const int n_steps = Tg * RESOLUTION;   // 110
  const float dt = 1.0f / (float)n_steps;
  for (int s = 0; s < n_steps; ++s) {
    edge_kernel<<<(E + 255) / 256, 256, 0, stream>>>(xs, edge_w, src, dst, msg, E, N, B);
    node_kernel<<<(B * N + 255) / 256, 256, 0, stream>>>(xs, msg, wdeg, gate, jt,
                                                         diff_a, react_r, B, N, T, s, dt);
  }
}
